// CausalSelfAttention_87883620811407
// MI455X (gfx1250) — matrix operand order repaired and checked by reading, not compiled, not run on hardware
//
#include <hip/hip_runtime.h>
#include <hip/hip_bf16.h>

// ---------------------------------------------------------------------------
// CausalSelfAttention (GQA + RoPE + QK-RMSNorm) for MI455X / gfx1250.
// B=2 T=2048 C=2048 H=16 KVH=4 HD=128.  ~155 GFLOP, compute bound ->
// everything routed through v_wmma_f32_16x16x32_f16 (wave32 WMMA), with
// gfx1250 async global->LDS copies (ASYNCcnt) for double-buffered staging.
// ---------------------------------------------------------------------------

#define BB   2
#define TT   2048
#define CC   2048
#define HH   16
#define KVHH 4
#define HDD  128
#define BT   (BB * TT)      // 4096
#define QD   (HH * HDD)     // 2048
#define KVD  (KVHH * HDD)   // 512

typedef __attribute__((ext_vector_type(16))) _Float16 v16h;
typedef __attribute__((ext_vector_type(8)))  _Float16 v8h;
typedef __attribute__((ext_vector_type(4)))  _Float16 v4h;
typedef __attribute__((ext_vector_type(8)))  float    v8f;
typedef __attribute__((ext_vector_type(4)))  int      v4i;

#define AS1 __attribute__((address_space(1)))
#define AS3 __attribute__((address_space(3)))

__device__ __forceinline__ v8f zero8() {
  v8f z = {0.f, 0.f, 0.f, 0.f, 0.f, 0.f, 0.f, 0.f};
  return z;
}

__device__ __forceinline__ v8f wmma_f16(v16h a, v16h b, v8f c) {
  // D = A(16x32 f16) * B(32x16 f16) + C(16x16 f32)
  return __builtin_amdgcn_wmma_f32_16x16x32_f16(
      false, a, false, b, (short)0, c, false, false);
}

// 16B global -> LDS async copy (GLOBAL_LOAD_ASYNC_TO_LDS_B128, ASYNCcnt).
// Builtin signature: void(v4i AS1* gsrc, v4i AS3* ldst, imm offset, imm cpol).
__device__ __forceinline__ void async_cp16(const _Float16* g, _Float16* l) {
#if __has_builtin(__builtin_amdgcn_global_load_async_to_lds_b128)
  __builtin_amdgcn_global_load_async_to_lds_b128(
      (AS1 v4i*)g, (AS3 v4i*)l, 0, 0);
#else
  *(uint4*)l = *(const uint4*)g;
#endif
}

__device__ __forceinline__ void wait_async0() {
#if __has_builtin(__builtin_amdgcn_s_wait_asynccnt)
  __builtin_amdgcn_s_wait_asynccnt(0);
#else
  asm volatile("s_wait_asynccnt 0x0" ::: "memory");
#endif
}

// A-matrix 16x32 f16 fragment from row-major LDS tile [16+][ld] (ld in halves,
// multiple of 8). Lane l holds row (l&15); K runs [kb,kb+8) and [kb+16,kb+24),
// kb = (l>>4)*8.  (CDNA5 ISA "16-bit A-Matrix 16x32")
__device__ __forceinline__ v16h load_A_frag(const _Float16* base, int ld) {
  const int lane = threadIdx.x & 31;
  const _Float16* p = base + (lane & 15) * ld + ((lane >> 4) << 3);
  v8h lo = *(const v8h*)(p);
  v8h hi = *(const v8h*)(p + 16);
  v16h a;
#pragma unroll
  for (int i = 0; i < 8; ++i) { a[i] = lo[i]; a[i + 8] = hi[i]; }
  return a;
}

// B-matrix 32x16 f16 fragment from LDS tile holding B^T row-major:
// Bt[n][k], stride ld halves. Lane l holds column n=(l&15); K runs [kb,kb+8)
// and [kb+16,kb+24), kb = (l>>4)*8.
__device__ __forceinline__ v16h load_B_frag(const _Float16* base, int ld) {
  const int lane = threadIdx.x & 31;
  const _Float16* p = base + (lane & 15) * ld + ((lane >> 4) << 3);
  v8h lo = *(const v8h*)(p);
  v8h hi = *(const v8h*)(p + 16);
  v16h b;
#pragma unroll
  for (int i = 0; i < 8; ++i) { b[i] = lo[i]; b[i + 8] = hi[i]; }
  return b;
}

__device__ __forceinline__ float red_max16(float v) {
  v = fmaxf(v, __shfl_xor(v, 1, 32));
  v = fmaxf(v, __shfl_xor(v, 2, 32));
  v = fmaxf(v, __shfl_xor(v, 4, 32));
  v = fmaxf(v, __shfl_xor(v, 8, 32));
  return v;
}
__device__ __forceinline__ float red_sum16(float v) {
  v += __shfl_xor(v, 1, 32);
  v += __shfl_xor(v, 2, 32);
  v += __shfl_xor(v, 4, 32);
  v += __shfl_xor(v, 8, 32);
  return v;
}
__device__ __forceinline__ float red_sum32(float v) {
  v += __shfl_xor(v, 1, 32);
  v += __shfl_xor(v, 2, 32);
  v += __shfl_xor(v, 4, 32);
  v += __shfl_xor(v, 8, 32);
  v += __shfl_xor(v, 16, 32);
  return v;
}

// ---------------------------------------------------------------------------
// GEMM: C[M,N] = A[M,K] * B[K,N].  A f32 (converted) or f16 (async-staged);
// B f32 converted while staging transposed. 128x128x32 tiles, 256 thr = 8
// waves (4m x 2n), each wave a 32x64 sub-tile = 2x4 WMMA accumulators.
// CMODE: 0 = f32 row-major, 1 = f16 row-major, 2 = f16 V-transposed layout.
// ---------------------------------------------------------------------------
#define GBM 128
#define GBN 128
#define GBK 32
#define GLD 40   // padded LDS row stride in halves (32 + 8)

template <bool A_HALF, int CMODE>
__global__ __launch_bounds__(256) void gemm_wmma(const void* __restrict__ Av,
                                                 const float* __restrict__ Bmat,
                                                 void* __restrict__ Cv,
                                                 int M, int N, int K) {
  __shared__ _Float16 As[GBM * GLD];
  __shared__ _Float16 Bs[GBN * GLD];

  const int tid  = threadIdx.x;
  const int lane = tid & 31;
  const int wave = tid >> 5;
  const int wm   = wave & 3;   // 0..3 -> 32-row strip
  const int wn   = wave >> 2;  // 0..1 -> 64-col strip
  const int m0 = blockIdx.y * GBM;
  const int n0 = blockIdx.x * GBN;

  v8f acc[2][4];
#pragma unroll
  for (int i = 0; i < 2; ++i)
#pragma unroll
    for (int j = 0; j < 4; ++j) acc[i][j] = zero8();

  for (int k0 = 0; k0 < K; k0 += GBK) {
    __syncthreads();
    // ---- stage A tile (rows m0..m0+127, k0..k0+31) ----
    {
      const int row = tid >> 1;
      const int cs  = (tid & 1) * 16;
      if (A_HALF) {
        const _Float16* src = (const _Float16*)Av + (size_t)(m0 + row) * K + k0 + cs;
        async_cp16(src,     &As[row * GLD + cs]);
        async_cp16(src + 8, &As[row * GLD + cs + 8]);
      } else {
        const float* src = (const float*)Av + (size_t)(m0 + row) * K + k0 + cs;
#pragma unroll
        for (int j = 0; j < 4; ++j) {
          float4 f = *(const float4*)(src + j * 4);
          v4h h;
          h[0] = (_Float16)f.x; h[1] = (_Float16)f.y;
          h[2] = (_Float16)f.z; h[3] = (_Float16)f.w;
          *(v4h*)&As[row * GLD + cs + j * 4] = h;
        }
      }
    }
    // ---- stage B tile transposed: Bs[n][k] = B[k0+k][n0+n] ----
    {
      const int kr = tid >> 3;         // 0..31
      const int ns = (tid & 7) * 16;   // 0..112
      const float* src = Bmat + (size_t)(k0 + kr) * N + n0 + ns;
#pragma unroll
      for (int j = 0; j < 4; ++j) {
        float4 f = *(const float4*)(src + j * 4);
        Bs[(ns + j * 4 + 0) * GLD + kr] = (_Float16)f.x;
        Bs[(ns + j * 4 + 1) * GLD + kr] = (_Float16)f.y;
        Bs[(ns + j * 4 + 2) * GLD + kr] = (_Float16)f.z;
        Bs[(ns + j * 4 + 3) * GLD + kr] = (_Float16)f.w;
      }
      if (k0 + GBK < K)  // prefetch next B tile into L2/WGP$
        __builtin_prefetch(src + (size_t)GBK * N, 0, 1);
    }
    if (A_HALF) wait_async0();
    __syncthreads();
    // ---- compute ----
    const _Float16* Abase = &As[(wm * 32) * GLD];
    v16h a0 = load_A_frag(Abase, GLD);
    v16h a1 = load_A_frag(Abase + 16 * GLD, GLD);
#pragma unroll
    for (int j = 0; j < 4; ++j) {
      v16h b = load_B_frag(&Bs[(wn * 64 + j * 16) * GLD], GLD);
      acc[0][j] = wmma_f16(a0, b, acc[0][j]);
      acc[1][j] = wmma_f16(a1, b, acc[1][j]);
    }
  }

  // ---- epilogue: C/D layout = (row r+8*(lane>>4), col lane&15) per VGPR r ----
  const int col = lane & 15;
  const int hi  = lane >> 4;
#pragma unroll
  for (int i = 0; i < 2; ++i)
#pragma unroll
    for (int j = 0; j < 4; ++j)
#pragma unroll
      for (int r = 0; r < 8; ++r) {
        const int m = m0 + wm * 32 + i * 16 + r + 8 * hi;
        const int n = n0 + wn * 64 + j * 16 + col;
        const float val = acc[i][j][r];
        if (CMODE == 0) {
          ((float*)Cv)[(size_t)m * N + n] = val;
        } else if (CMODE == 1) {
          ((_Float16*)Cv)[(size_t)m * N + n] = (_Float16)val;
        } else {
          // V stored transposed: vT[(b*KVD + n) * T + t],  m = b*T + t
          const int bb = m >> 11;      // / TT
          const int t  = m & (TT - 1);
          ((_Float16*)Cv)[((size_t)bb * KVD + n) * TT + t] = (_Float16)val;
        }
      }
}

// ---------------------------------------------------------------------------
// RoPE + RMSNorm for Q and K (f16 in, f16 out). One block per (b,t) row,
// 4 waves; lane l covers dims l, l+32 and their rotation partners (+64).
// ---------------------------------------------------------------------------
__global__ __launch_bounds__(128) void rope_rmsnorm(const _Float16* __restrict__ qraw,
                                                    const _Float16* __restrict__ kraw,
                                                    const float* __restrict__ cosb,
                                                    const float* __restrict__ sinb,
                                                    _Float16* __restrict__ qh,
                                                    _Float16* __restrict__ kh) {
  const int row  = blockIdx.x;       // b*T + t
  const int t    = row & (TT - 1);
  const int wave = threadIdx.x >> 5;
  const int lane = threadIdx.x & 31;
  const int d0 = lane, d1 = lane + 32;
  const float c0 = cosb[t * 64 + d0], s0 = sinb[t * 64 + d0];
  const float c1 = cosb[t * 64 + d1], s1 = sinb[t * 64 + d1];

  for (int h = wave; h < HH; h += 4) {
    const _Float16* src = qraw + (size_t)row * QD + h * HDD;
    float x1a = (float)src[d0], x2a = (float)src[d0 + 64];
    float x1b = (float)src[d1], x2b = (float)src[d1 + 64];
    float oa0 =  x1a * c0 + x2a * s0;
    float oa1 = -x1a * s0 + x2a * c0;
    float ob0 =  x1b * c1 + x2b * s1;
    float ob1 = -x1b * s1 + x2b * c1;
    float ss  = red_sum32(oa0 * oa0 + oa1 * oa1 + ob0 * ob0 + ob1 * ob1);
    float sc  = rsqrtf(ss * (1.0f / 128.0f) + 1e-5f);
    _Float16* dst = qh + (size_t)row * QD + h * HDD;
    dst[d0]      = (_Float16)(oa0 * sc);
    dst[d0 + 64] = (_Float16)(oa1 * sc);
    dst[d1]      = (_Float16)(ob0 * sc);
    dst[d1 + 64] = (_Float16)(ob1 * sc);
  }
  if (wave < KVHH) {
    const int h = wave;
    const _Float16* src = kraw + (size_t)row * KVD + h * HDD;
    float x1a = (float)src[d0], x2a = (float)src[d0 + 64];
    float x1b = (float)src[d1], x2b = (float)src[d1 + 64];
    float oa0 =  x1a * c0 + x2a * s0;
    float oa1 = -x1a * s0 + x2a * c0;
    float ob0 =  x1b * c1 + x2b * s1;
    float ob1 = -x1b * s1 + x2b * c1;
    float ss  = red_sum32(oa0 * oa0 + oa1 * oa1 + ob0 * ob0 + ob1 * ob1);
    float sc  = rsqrtf(ss * (1.0f / 128.0f) + 1e-5f);
    _Float16* dst = kh + (size_t)row * KVD + h * HDD;
    dst[d0]      = (_Float16)(oa0 * sc);
    dst[d0 + 64] = (_Float16)(oa1 * sc);
    dst[d1]      = (_Float16)(ob0 * sc);
    dst[d1 + 64] = (_Float16)(ob1 * sc);
  }
}

// ---------------------------------------------------------------------------
// Flash attention (causal, GQA 4:1). Block = 128 thr = 4 waves, 64 query rows
// (16/wave), 32-key tiles. K/V tiles double-buffered via async global->LDS
// copies (next tile in flight while computing current). V pre-transposed in
// global memory so Vt staging is contiguous. P re-laid-out C-frag -> A-frag
// through per-wave LDS.
// ---------------------------------------------------------------------------
#define FBQ 64
#define FBK 32
#define LQK 136  // 128 + 8 pad, halves
#define LVT 40   // 32 + 8 pad
#define LPS 40

__global__ __launch_bounds__(128) void flash_attn(const _Float16* __restrict__ qh,
                                                  const _Float16* __restrict__ kh,
                                                  const _Float16* __restrict__ vT,
                                                  _Float16* __restrict__ yh) {
  __shared__ _Float16 Qs[FBQ * LQK];
  __shared__ _Float16 Ks[2][FBK * LQK];
  __shared__ _Float16 Vt[2][HDD * LVT];    // Vt[hd][key]
  __shared__ _Float16 Ps[4 * 16 * LPS];    // per-wave P staging

  const int m0  = blockIdx.x * FBQ;
  const int h   = blockIdx.y;
  const int b   = blockIdx.z;
  const int kvh = h >> 2;                  // n_rep = 4
  const int tid  = threadIdx.x;
  const int wave = tid >> 5;
  const int lane = tid & 31;
  const int colk = lane & 15;
  const int hi   = lane >> 4;

  auto issue_tiles = [&](int j0, int buf) {
    {  // K tile: [key][hd] row-major (== B^T for S = Q*K^T)
      const int r  = tid >> 2;             // 0..31
      const int cs = (tid & 3) * 32;
      const _Float16* src = kh + ((size_t)(b * TT + j0 + r) * KVHH + kvh) * HDD + cs;
      _Float16* dst = &Ks[buf][r * LQK + cs];
#pragma unroll
      for (int j = 0; j < 4; ++j) async_cp16(src + j * 8, dst + j * 8);
    }
    {  // V tile: Vt[hd][key]; vT rows are contiguous along t
      const _Float16* src = vT + ((size_t)(b * KVD + kvh * HDD + tid)) * TT + j0;
      _Float16* dst = &Vt[buf][tid * LVT];
#pragma unroll
      for (int j = 0; j < 4; ++j) async_cp16(src + j * 8, dst + j * 8);
    }
  };

  // ---- stage Q tile (async) + first K/V tile ----
  {
    const int r  = tid >> 1;
    const int cs = (tid & 1) * 64;
    const _Float16* src = qh + ((size_t)(b * TT + m0 + r) * HH + h) * HDD + cs;
    _Float16* dst = &Qs[r * LQK + cs];
#pragma unroll
    for (int j = 0; j < 8; ++j) async_cp16(src + j * 8, dst + j * 8);
  }
  issue_tiles(0, 0);
  wait_async0();
  __syncthreads();

  // Q fragments for this wave's 16 rows, all 4 K-chunks of HD=128
  v16h qf[4];
#pragma unroll
  for (int kc = 0; kc < 4; ++kc)
    qf[kc] = load_A_frag(&Qs[(wave * 16) * LQK + kc * 32], LQK);

  float mrow[8], lrow[8];
  v8f o[8];
#pragma unroll
  for (int r = 0; r < 8; ++r) { mrow[r] = -1e30f; lrow[r] = 0.f; }
#pragma unroll
  for (int c = 0; c < 8; ++c) o[c] = zero8();

  const float scale = 0.08838834764831845f;  // 1/sqrt(128)
  const int jend = m0 + FBQ;                 // causal: keys < m0+64

  int it = 0;
  for (int j0 = 0; j0 < jend; j0 += FBK, ++it) {
    const int buf = it & 1;
    if (it) {          // tiles for this iter were issued last iter
      wait_async0();
      __syncthreads();
    }
    if (j0 + FBK < jend) issue_tiles(j0 + FBK, buf ^ 1);  // overlap next tile

    // ---- S = Q * K^T (16 x 32 per wave, two 16-col fragments) ----
    v8f s0 = zero8(), s1 = zero8();
#pragma unroll
    for (int kc = 0; kc < 4; ++kc) {
      v16h b0 = load_B_frag(&Ks[buf][0 * LQK + kc * 32], LQK);
      v16h b1 = load_B_frag(&Ks[buf][16 * LQK + kc * 32], LQK);
      s0 = wmma_f16(qf[kc], b0, s0);
      s1 = wmma_f16(qf[kc], b1, s1);
    }

    // ---- online softmax (per row, 16-lane shfl reductions) ----
#pragma unroll
    for (int r = 0; r < 8; ++r) {
      const int m = m0 + wave * 16 + r + 8 * hi;
      float v0 = (j0 + colk      <= m) ? s0[r] * scale : -1e30f;
      float v1 = (j0 + 16 + colk <= m) ? s1[r] * scale : -1e30f;
      float mn = fmaxf(mrow[r], red_max16(fmaxf(v0, v1)));
      float e0 = __expf(v0 - mn);
      float e1 = __expf(v1 - mn);
      float alpha = __expf(mrow[r] - mn);
      mrow[r] = mn;
      lrow[r] = lrow[r] * alpha + red_sum16(e0 + e1);
#pragma unroll
      for (int c = 0; c < 8; ++c) o[c][r] *= alpha;
      _Float16* pp = &Ps[(wave * 16 + r + 8 * hi) * LPS];
      pp[colk]      = (_Float16)e0;
      pp[16 + colk] = (_Float16)e1;
    }
    __syncthreads();  // Ps visible for A-fragment reads

    // ---- O += P * V ----
    v16h pa = load_A_frag(&Ps[(wave * 16) * LPS], LPS);
#pragma unroll
    for (int c = 0; c < 8; ++c) {
      v16h bv = load_B_frag(&Vt[buf][(c * 16) * LVT], LVT);
      o[c] = wmma_f16(pa, bv, o[c]);
    }
  }

  // ---- finalize: O /= l, write y[b,t,h,:] as f16 ----
  float inv[8];
#pragma unroll
  for (int r = 0; r < 8; ++r) inv[r] = 1.0f / lrow[r];
#pragma unroll
  for (int c = 0; c < 8; ++c)
#pragma unroll
    for (int r = 0; r < 8; ++r) {
      const int trow = m0 + wave * 16 + r + 8 * hi;
      const int col  = h * HDD + c * 16 + colk;
      yh[(size_t)(b * TT + trow) * QD + col] = (_Float16)(o[c][r] * inv[r]);
    }
}

// ---------------------------------------------------------------------------
extern "C" void kernel_launch(void* const* d_in, const int* in_sizes, int n_in,
                              void* d_out, int out_size, void* d_ws, size_t ws_size,
                              hipStream_t stream) {
  (void)in_sizes; (void)n_in; (void)out_size; (void)ws_size;
  const float* x    = (const float*)d_in[0];
  const float* cosb = (const float*)d_in[1];
  const float* sinb = (const float*)d_in[2];
  const float* wq   = (const float*)d_in[3];
  const float* wk   = (const float*)d_in[4];
  const float* wv   = (const float*)d_in[5];
  const float* wo   = (const float*)d_in[6];
  // d_in[7] = window_size (0 => full causal), unused.

  char* ws = (char*)d_ws;
  size_t off = 0;
  auto alloc = [&](size_t bytes) -> void* {
    void* p = ws + off;
    off = (off + bytes + 255) & ~(size_t)255;
    return p;
  };
  _Float16* qraw = (_Float16*)alloc((size_t)BT * QD  * 2);  // pre-rope q
  _Float16* kraw = (_Float16*)alloc((size_t)BT * KVD * 2);  // pre-rope k
  _Float16* qn   = (_Float16*)alloc((size_t)BT * QD  * 2);  // normed q
  _Float16* kn   = (_Float16*)alloc((size_t)BT * KVD * 2);  // normed k
  _Float16* vTb  = (_Float16*)alloc((size_t)BT * KVD * 2);  // v, transposed
  _Float16* yh   = qraw;  // qraw dead after rope_rmsnorm -> reuse for y

  dim3 blk(256);
  // QKV projections (x f32 converted while staging; outputs f16)
  gemm_wmma<false, 1><<<dim3(QD / GBN,  BT / GBM), blk, 0, stream>>>(x, wq, qraw, BT, QD,  CC);
  gemm_wmma<false, 1><<<dim3(KVD / GBN, BT / GBM), blk, 0, stream>>>(x, wk, kraw, BT, KVD, CC);
  gemm_wmma<false, 2><<<dim3(KVD / GBN, BT / GBM), blk, 0, stream>>>(x, wv, vTb,  BT, KVD, CC);

  // RoPE + RMSNorm on q,k
  rope_rmsnorm<<<BT, 128, 0, stream>>>(qraw, kraw, cosb, sinb, qn, kn);

  // Causal flash attention -> y (f16)
  flash_attn<<<dim3(TT / FBQ, HH, BB), 128, 0, stream>>>(qn, kn, vTb, yh);

  // Output projection: out = y @ wo (f32 out)
  gemm_wmma<true, 0><<<dim3(CC / GBN, BT / GBM), blk, 0, stream>>>(yh, wo, d_out, BT, CC, CC);
}
